// R2P2_29867202577009
// MI455X (gfx1250) — matrix operand order repaired and checked by reading, not compiled, not run on hardware
//
#include <hip/hip_runtime.h>
#include <hip/hip_bf16.h>
#include <math.h>
#include <stdint.h>

typedef __attribute__((ext_vector_type(16))) _Float16 v16h;
typedef __attribute__((ext_vector_type(8)))  _Float16 v8h;
typedef __attribute__((ext_vector_type(8)))  float    v8f;
typedef __attribute__((ext_vector_type(4)))  unsigned int u32x4;
typedef __attribute__((ext_vector_type(4)))  int          i32x4;
typedef __attribute__((ext_vector_type(8)))  int          i32x8;

#define BB   256
#define TP   20
#define TF   30
#define GH   100
#define GW   100
#define HID  32
#define NPOS (BB * GH * GW)   // 2,560,000 output positions per layer

// ---------------------------------------------------------------------------
// f32 -> f16 elementwise convert (lidar input)
// ---------------------------------------------------------------------------
__global__ void cvt_f16_kernel(const float* __restrict__ src,
                               _Float16* __restrict__ dst, int n) {
    int i = blockIdx.x * 256 + threadIdx.x;
    if (i < n) dst[i] = (_Float16)src[i];
}

// ---------------------------------------------------------------------------
// Conv weight prep: (2,2,CIN,32) HWIO f32  ->  wT[n][k] f16, K zero-padded
// ---------------------------------------------------------------------------
__global__ void prep_wT_kernel(const float* __restrict__ w,
                               _Float16* __restrict__ wT, int KTOT, int KPAD) {
    int i = blockIdx.x * 256 + threadIdx.x;
    if (i >= 32 * KPAD) return;
    int n = i / KPAD, k = i % KPAD;
    wT[i] = (k < KTOT) ? (_Float16)w[k * 32 + n] : (_Float16)0.f;
}

// ---------------------------------------------------------------------------
// Collapse the linear MLP:  Wc = W2 @ W1 (4x32),  bc = W2 @ b1 + b2 (4)
// ---------------------------------------------------------------------------
__global__ void collapse_mlp_kernel(const float* __restrict__ w1,
                                    const float* __restrict__ b1,
                                    const float* __restrict__ w2,
                                    const float* __restrict__ b2,
                                    float* __restrict__ wc,
                                    float* __restrict__ bc) {
    int t = threadIdx.x;
    if (t < 128) {
        int o = t >> 5, k = t & 31;
        float acc = 0.f;
        for (int n = 0; n < 512; ++n) acc += w2[o * 512 + n] * w1[n * 32 + k];
        wc[o * 32 + k] = acc;
    } else if (t < 132) {
        int o = t - 128;
        float acc = b2[o];
        for (int n = 0; n < 512; ++n) acc += w2[o * 512 + n] * b1[n];
        bc[o] = acc;
    }
}

// ---------------------------------------------------------------------------
// TDM helper: stage a (32 x KPAD) f16 weight panel into LDS with one
// tensor_load_to_lds (D# per ISA 8.3/8.4), waiting on TENSORcnt.
// Falls back to a cooperative vector copy when the builtin is unavailable.
// ---------------------------------------------------------------------------
__device__ __forceinline__ void stage_weights_lds(_Float16* lds_dst,
                                                  const _Float16* gsrc,
                                                  int n_halfs /* 32*KPAD */,
                                                  int tid) {
#if defined(__HIP_DEVICE_COMPILE__) && __has_builtin(__builtin_amdgcn_tensor_load_to_lds)
    if (tid < 32) {  // one wave issues the DMA (EXEC ignored by TDM)
        const unsigned nd      = (unsigned)(n_halfs / 2);            // 4B elements
        const unsigned lds_off = (unsigned)(uintptr_t)lds_dst;       // LDS generic addr low32 = offset
        const unsigned long long ga = (unsigned long long)(uintptr_t)gsrc;

        u32x4 g0;
        g0[0] = 1u;                                                  // count=1, user mode
        g0[1] = lds_off;                                             // lds_addr [63:32]
        g0[2] = (unsigned)(ga & 0xffffffffu);                        // global_addr [95:64]
        g0[3] = (unsigned)((ga >> 32) & 0x01ffffffu) | (2u << 30);   // addr[56:32] | type=2

        i32x8 g1 = {};
        g1[0] = (2 << 16);                 // data_size = 4B; no multicast/pad/iterate
        g1[1] = (int)((nd & 0xffffu) << 16);           // tensor_dim0[15:0] @ bits 63:48
        g1[2] = (int)((nd >> 16) | (1u << 16));        // tensor_dim0[31:16] | tensor_dim1=1
        g1[3] = (int)((nd & 0xffffu) << 16);           // tensor_dim1 hi=0 | tile_dim0 = nd
        g1[4] = 1;                                     // tile_dim1 = 1, tile_dim2 = 0
        g1[5] = (int)nd;                               // tensor_dim0_stride lo
        g1[6] = 0;                                     // stride0 hi | stride1 lo
        g1[7] = 0;

        i32x4 g2 = {}, g3 = {};
#if defined(__clang_major__) && (__clang_major__ >= 23)
        i32x8 g4 = {};
        __builtin_amdgcn_tensor_load_to_lds(g0, g1, g2, g3, g4, 0);
#else
        __builtin_amdgcn_tensor_load_to_lds(g0, g1, g2, g3, 0);
#endif
#if __has_builtin(__builtin_amdgcn_s_wait_tensorcnt)
        __builtin_amdgcn_s_wait_tensorcnt(0);
#else
        asm volatile("s_wait_tensorcnt 0x0" ::: "memory");
#endif
    }
    __syncthreads();
#else
    for (int i = tid; i < n_halfs / 8; i += 256)
        ((v8h*)lds_dst)[i] = ((const v8h*)gsrc)[i];
    __syncthreads();
#endif
}

// ---------------------------------------------------------------------------
// 2x2 conv (pad bottom/right) + bias + ReLU as implicit GEMM via WMMA.
// Each wave: TWO 16-position M-tiles x 32 cout (B fragments reused) -> 4 WMMA
// per k-step.  8 waves / 256-thr block, weights staged in LDS by TDM.
// ---------------------------------------------------------------------------
template <int CIN>
__global__ __launch_bounds__(256) void conv2x2_wmma_kernel(
    const _Float16* __restrict__ fin,   // (B,100,100,CIN) f16
    const _Float16* __restrict__ wT,    // (32, KPAD) f16
    const float* __restrict__ bias,     // (32) f32
    _Float16* __restrict__ fout) {      // (B,100,100,32) f16
    constexpr int KTOT   = 4 * CIN;
    constexpr int KSTEPS = (KTOT + 31) / 32;
    constexpr int KPAD   = KSTEPS * 32;

    __shared__ __align__(16) _Float16 lw[32 * KPAD];
    stage_weights_lds(lw, wT, 32 * KPAD, threadIdx.x);

    const int wave  = threadIdx.x >> 5;
    const int lane  = threadIdx.x & 31;
    const int group = blockIdx.x * 8 + wave;   // 32 positions per wave
    const int mrow  = lane & 15;
    const int hi    = lane >> 4;
    const int n0    = lane & 15;

    const int pos0 = group * 32 + mrow;        // M-tile 0 row
    const int pos1 = pos0 + 16;                // M-tile 1 row
    const int img0 = pos0 / (GH * GW), rem0 = pos0 % (GH * GW);
    const int img1 = pos1 / (GH * GW), rem1 = pos1 % (GH * GW);
    const int y0 = rem0 / GW, x0 = rem0 % GW;
    const int y1 = rem1 / GW, x1 = rem1 % GW;

    v8f c00 = {}, c01 = {}, c10 = {}, c11 = {};
#pragma unroll
    for (int ks = 0; ks < KSTEPS; ++ks) {
        v16h a0 = {}, a1 = {};
        if constexpr (CIN == 32) {
            const int dy = ks >> 1, dx = ks & 1;
            {   // M-tile 0
                const int yy = y0 + dy, xx = x0 + dx;
                v8h lo = {}, hh = {};
                if (yy < GH && xx < GW) {
                    const _Float16* p = fin + (((img0 * GH + yy) * GW + xx) * 32 + hi * 16);
                    lo = *(const v8h*)p;
                    hh = *(const v8h*)(p + 8);
                }
#pragma unroll
                for (int j = 0; j < 8; ++j) { a0[j] = lo[j]; a0[8 + j] = hh[j]; }
            }
            {   // M-tile 1
                const int yy = y1 + dy, xx = x1 + dx;
                v8h lo = {}, hh = {};
                if (yy < GH && xx < GW) {
                    const _Float16* p = fin + (((img1 * GH + yy) * GW + xx) * 32 + hi * 16);
                    lo = *(const v8h*)p;
                    hh = *(const v8h*)(p + 8);
                }
#pragma unroll
                for (int j = 0; j < 8; ++j) { a1[j] = lo[j]; a1[8 + j] = hh[j]; }
            }
        } else {
            // CIN==2: KTOT=8 padded to 32; nonzero only lanes 0-15, elems 0-7
            if (hi == 0) {
#pragma unroll
                for (int tap = 0; tap < 4; ++tap) {
                    const int dy = tap >> 1, dx = tap & 1;
                    {
                        const int yy = y0 + dy, xx = x0 + dx;
                        if (yy < GH && xx < GW) {
                            const _Float16* p = fin + (((img0 * GH + yy) * GW + xx) * 2);
                            a0[tap * 2] = p[0]; a0[tap * 2 + 1] = p[1];
                        }
                    }
                    {
                        const int yy = y1 + dy, xx = x1 + dx;
                        if (yy < GH && xx < GW) {
                            const _Float16* p = fin + (((img1 * GH + yy) * GW + xx) * 2);
                            a1[tap * 2] = p[0]; a1[tap * 2 + 1] = p[1];
                        }
                    }
                }
            }
        }
        const int kb = ks * 32 + hi * 16;  // B frag: lane col n, K = kb..kb+15
        v16h b0 = *(const v16h*)&lw[n0 * KPAD + kb];
        v16h b1 = *(const v16h*)&lw[(n0 + 16) * KPAD + kb];
        c00 = __builtin_amdgcn_wmma_f32_16x16x32_f16(false, a0, false, b0, (short)0, c00, false, false);
        c01 = __builtin_amdgcn_wmma_f32_16x16x32_f16(false, a0, false, b1, (short)0, c01, false, false);
        c10 = __builtin_amdgcn_wmma_f32_16x16x32_f16(false, a1, false, b0, (short)0, c10, false, false);
        c11 = __builtin_amdgcn_wmma_f32_16x16x32_f16(false, a1, false, b1, (short)0, c11, false, false);
    }

    const float bia0 = bias[n0], bia1 = bias[n0 + 16];
#pragma unroll
    for (int r = 0; r < 8; ++r) {
        const int m = hi * 8 + r;
        {
            const int opos = group * 32 + m;
            float v0 = c00[r] + bia0; v0 = v0 > 0.f ? v0 : 0.f;
            float v1 = c01[r] + bia1; v1 = v1 > 0.f ? v1 : 0.f;
            fout[opos * 32 + n0]      = (_Float16)v0;
            fout[opos * 32 + n0 + 16] = (_Float16)v1;
        }
        {
            const int opos = group * 32 + 16 + m;
            float v0 = c10[r] + bia0; v0 = v0 > 0.f ? v0 : 0.f;
            float v1 = c11[r] + bia1; v1 = v1 > 0.f ? v1 : 0.f;
            fout[opos * 32 + n0]      = (_Float16)v0;
            fout[opos * 32 + n0 + 16] = (_Float16)v1;
        }
    }
}

// ---------------------------------------------------------------------------
__device__ __forceinline__ float sigmoidf_(float x) { return 1.f / (1.f + expf(-x)); }
__device__ __forceinline__ float softplusf_(float x) { return x > 20.f ? x : log1pf(expf(x)); }

// ---------------------------------------------------------------------------
// GRU encoder over 20 past steps. One 1024-thread block; (b,j) pairs spread
// 8-per-thread; h state + weights in LDS.
// ---------------------------------------------------------------------------
__global__ __launch_bounds__(1024) void enc_gru_kernel(
    const float* __restrict__ past,  // (B,20,2)
    const float* __restrict__ wih,   // (96,2)
    const float* __restrict__ whh,   // (96,32)
    const float* __restrict__ bih,   // (96)
    const float* __restrict__ bhh,   // (96)
    float* __restrict__ h_out) {     // (B,32)
    __shared__ float swih[96 * 2];
    __shared__ float swhh[96 * 32];
    __shared__ float sbih[96], sbhh[96];
    __shared__ float sh[BB * HID];

    for (int i = threadIdx.x; i < 96 * 2; i += 1024) swih[i] = wih[i];
    for (int i = threadIdx.x; i < 96 * 32; i += 1024) swhh[i] = whh[i];
    for (int i = threadIdx.x; i < 96; i += 1024) { sbih[i] = bih[i]; sbhh[i] = bhh[i]; }
    for (int i = threadIdx.x; i < BB * HID; i += 1024) sh[i] = 0.f;
    __syncthreads();

    for (int t = 0; t < TP; ++t) {
        float hn[8];
#pragma unroll
        for (int q = 0; q < 8; ++q) {
            const int idx = threadIdx.x + q * 1024;
            const int b = idx >> 5, j = idx & 31;
            const float x0 = past[b * (TP * 2) + t * 2];
            const float x1 = past[b * (TP * 2) + t * 2 + 1];
            const float* hrow = &sh[b * HID];
            float gr = sbih[j]      + swih[j * 2] * x0        + swih[j * 2 + 1] * x1;
            float gz = sbih[j + 32] + swih[(j + 32) * 2] * x0 + swih[(j + 32) * 2 + 1] * x1;
            float gn = sbih[j + 64] + swih[(j + 64) * 2] * x0 + swih[(j + 64) * 2 + 1] * x1;
            float hr = sbhh[j], hz = sbhh[j + 32], hhn = sbhh[j + 64];
            for (int k = 0; k < HID; ++k) {
                const float hk = hrow[k];
                hr  += swhh[j * HID + k] * hk;
                hz  += swhh[(j + 32) * HID + k] * hk;
                hhn += swhh[(j + 64) * HID + k] * hk;
            }
            const float r = sigmoidf_(gr + hr);
            const float z = sigmoidf_(gz + hz);
            const float n = tanhf(gn + r * hhn);
            hn[q] = (1.f - z) * n + z * hrow[j];
        }
        __syncthreads();
#pragma unroll
        for (int q = 0; q < 8; ++q) {
            const int idx = threadIdx.x + q * 1024;
            sh[(idx >> 5) * HID + (idx & 31)] = hn[q];
        }
        __syncthreads();
    }
    for (int i = threadIdx.x; i < BB * HID; i += 1024) h_out[i] = sh[i];
}

// ---------------------------------------------------------------------------
// Decoder rollout: 30 serial steps of bilinear-lookup -> GRU -> collapsed
// 4x32 head -> autoregressive position update.  One 1024-thread block.
// ---------------------------------------------------------------------------
__global__ __launch_bounds__(1024) void dec_rollout_kernel(
    const float* __restrict__ zin,      // (B,30,2)
    const float* __restrict__ past,     // (B,20,2)
    const float* __restrict__ h_enc,    // (B,32) f32
    const _Float16* __restrict__ ctx,   // (B,100,100,32) f16
    const float* __restrict__ wih,      // (96,34)
    const float* __restrict__ whh,      // (96,32)
    const float* __restrict__ bih,      // (96)
    const float* __restrict__ bhh,      // (96)
    const float* __restrict__ wc,       // (4,32)
    const float* __restrict__ bc,       // (4)
    float* __restrict__ out) {          // y[15360] ++ logabsdet[256]
    __shared__ _Float16 swih[96 * 34];
    __shared__ _Float16 swhh[96 * 32];
    __shared__ float sbih[96], sbhh[96];
    __shared__ float swc[4 * 32], sbc[4];
    __shared__ _Float16 sh[BB * HID];
    __shared__ __align__(16) _Float16 sx[BB * 34];
    __shared__ float sy1[BB * 2], sy2[BB * 2];
    __shared__ float slog[BB];

    for (int i = threadIdx.x; i < 96 * 34; i += 1024) swih[i] = (_Float16)wih[i];
    for (int i = threadIdx.x; i < 96 * 32; i += 1024) swhh[i] = (_Float16)whh[i];
    for (int i = threadIdx.x; i < 96; i += 1024) { sbih[i] = bih[i]; sbhh[i] = bhh[i]; }
    for (int i = threadIdx.x; i < 128; i += 1024) swc[i] = wc[i];
    if (threadIdx.x < 4) sbc[threadIdx.x] = bc[threadIdx.x];
    for (int i = threadIdx.x; i < BB * HID; i += 1024) sh[i] = (_Float16)h_enc[i];
    if (threadIdx.x < BB) {
        const int b = threadIdx.x;
        sy1[b * 2]     = past[b * (TP * 2) + 19 * 2];
        sy1[b * 2 + 1] = past[b * (TP * 2) + 19 * 2 + 1];
        sy2[b * 2]     = past[b * (TP * 2) + 18 * 2];
        sy2[b * 2 + 1] = past[b * (TP * 2) + 18 * 2 + 1];
        slog[b] = 0.f;
    }
    __syncthreads();

    for (int t = 0; t < TF; ++t) {
        // ---- phase A: bilinear lookup, build x = [y1, interp] (f16) ----
        {
            const int b = threadIdx.x >> 2;      // 4 threads / batch elem
            const int g = threadIdx.x & 3;       // 8 channels each
            const float qy = sy1[b * 2], qx = sy1[b * 2 + 1];
            float fyf = floorf(qy); fyf = fminf(fmaxf(fyf, 0.f), (float)(GH - 2));
            float fxf = floorf(qx); fxf = fminf(fmaxf(fxf, 0.f), (float)(GW - 2));
            const int fy = (int)fyf, fx = (int)fxf;
            const float ay = fminf(fmaxf(qy - fyf, 0.f), 1.f);
            const float ax = fminf(fmaxf(qx - fxf, 0.f), 1.f);
            const _Float16* base = ctx + (((b * GH + fy) * GW + fx) * 32 + g * 8);
            v8h tl = *(const v8h*)base;
            v8h tr = *(const v8h*)(base + 32);
            v8h bl = *(const v8h*)(base + GW * 32);
            v8h br = *(const v8h*)(base + GW * 32 + 32);
#pragma unroll
            for (int j = 0; j < 8; ++j) {
                const float ftl = (float)tl[j], ftr = (float)tr[j];
                const float fbl = (float)bl[j], fbr = (float)br[j];
                const float top = ftl + ax * (ftr - ftl);
                const float bot = fbl + ax * (fbr - fbl);
                sx[b * 34 + 2 + g * 8 + j] = (_Float16)(top + ay * (bot - top));
            }
            if (g == 0) { sx[b * 34] = (_Float16)qy; sx[b * 34 + 1] = (_Float16)qx; }
        }
        __syncthreads();
        // ---- phase B: GRU gates, 8 (b,j) pairs per thread ----
        float hn[8];
#pragma unroll
        for (int q = 0; q < 8; ++q) {
            const int idx = threadIdx.x + q * 1024;
            const int b = idx >> 5, j = idx & 31;
            const _Float16* xr = &sx[b * 34];
            const _Float16* hr = &sh[b * HID];
            float gr = sbih[j], gz = sbih[j + 32], gn = sbih[j + 64];
            for (int k = 0; k < 34; ++k) {
                const float xv = (float)xr[k];
                gr += (float)swih[j * 34 + k] * xv;
                gz += (float)swih[(j + 32) * 34 + k] * xv;
                gn += (float)swih[(j + 64) * 34 + k] * xv;
            }
            float hrr = sbhh[j], hz = sbhh[j + 32], hhn = sbhh[j + 64];
            for (int k = 0; k < HID; ++k) {
                const float hv = (float)hr[k];
                hrr += (float)swhh[j * HID + k] * hv;
                hz  += (float)swhh[(j + 32) * HID + k] * hv;
                hhn += (float)swhh[(j + 64) * HID + k] * hv;
            }
            const float r = sigmoidf_(gr + hrr);
            const float z = sigmoidf_(gz + hz);
            const float n = tanhf(gn + r * hhn);
            hn[q] = (1.f - z) * n + z * (float)hr[j];
        }
        __syncthreads();
#pragma unroll
        for (int q = 0; q < 8; ++q) {
            const int idx = threadIdx.x + q * 1024;
            sh[(idx >> 5) * HID + (idx & 31)] = (_Float16)hn[q];
        }
        __syncthreads();
        // ---- phase C: collapsed head + autoregressive update ----
        if (threadIdx.x < BB) {
            const int b = threadIdx.x;
            float ls[4];
#pragma unroll
            for (int o = 0; o < 4; ++o) {
                float acc = sbc[o];
                for (int k = 0; k < HID; ++k) acc += swc[o * HID + k] * (float)sh[b * HID + k];
                ls[o] = acc;
            }
            const float s0 = softplusf_(ls[2]);
            const float s1 = softplusf_(ls[3]);
            const float z0 = zin[b * (TF * 2) + t * 2];
            const float z1 = zin[b * (TF * 2) + t * 2 + 1];
            const float ny0 = 2.f * sy1[b * 2]     - sy2[b * 2]     + ls[0] + s0 * z0;
            const float ny1 = 2.f * sy1[b * 2 + 1] - sy2[b * 2 + 1] + ls[1] + s1 * z1;
            out[b * (TF * 2) + t * 2]     = ny0;
            out[b * (TF * 2) + t * 2 + 1] = ny1;
            slog[b] += logf(s0) + logf(s1);
            sy2[b * 2] = sy1[b * 2]; sy2[b * 2 + 1] = sy1[b * 2 + 1];
            sy1[b * 2] = ny0;        sy1[b * 2 + 1] = ny1;
        }
        __syncthreads();
    }
    if (threadIdx.x < BB) out[BB * TF * 2 + threadIdx.x] = slog[threadIdx.x];
}

// ---------------------------------------------------------------------------
static inline size_t align_up(size_t x) { return (x + 255) & ~(size_t)255; }

extern "C" void kernel_launch(void* const* d_in, const int* in_sizes, int n_in,
                              void* d_out, int out_size, void* d_ws, size_t ws_size,
                              hipStream_t stream) {
    const float* z        = (const float*)d_in[0];
    const float* past     = (const float*)d_in[1];
    const float* lidar    = (const float*)d_in[2];
    const float* conv_w[4] = { (const float*)d_in[3], (const float*)d_in[5],
                               (const float*)d_in[7], (const float*)d_in[9] };
    const float* conv_b[4] = { (const float*)d_in[4], (const float*)d_in[6],
                               (const float*)d_in[8], (const float*)d_in[10] };
    const float* enc_wih = (const float*)d_in[11];
    const float* enc_whh = (const float*)d_in[12];
    const float* enc_bih = (const float*)d_in[13];
    const float* enc_bhh = (const float*)d_in[14];
    const float* dec_wih = (const float*)d_in[15];
    const float* dec_whh = (const float*)d_in[16];
    const float* dec_bih = (const float*)d_in[17];
    const float* dec_bhh = (const float*)d_in[18];
    const float* mlp_w1  = (const float*)d_in[19];
    const float* mlp_b1  = (const float*)d_in[20];
    const float* mlp_w2  = (const float*)d_in[21];
    const float* mlp_b2  = (const float*)d_in[22];
    float* out = (float*)d_out;

    // workspace carve-up
    char* ws = (char*)d_ws;
    size_t off = 0;
    _Float16* lidarH = (_Float16*)(ws + off); off = align_up(off + (size_t)NPOS * 2 * 2);
    _Float16* fmapA  = (_Float16*)(ws + off); off = align_up(off + (size_t)NPOS * 32 * 2);
    _Float16* fmapB  = (_Float16*)(ws + off); off = align_up(off + (size_t)NPOS * 32 * 2);
    _Float16* wT0    = (_Float16*)(ws + off); off = align_up(off + 32 * 32 * 2);
    _Float16* wT1    = (_Float16*)(ws + off); off = align_up(off + 32 * 128 * 2);
    _Float16* wT2    = (_Float16*)(ws + off); off = align_up(off + 32 * 128 * 2);
    _Float16* wT3    = (_Float16*)(ws + off); off = align_up(off + 32 * 128 * 2);
    float*    h_enc  = (float*)(ws + off);    off = align_up(off + BB * HID * 4);
    float*    wc     = (float*)(ws + off);    off = align_up(off + 4 * 32 * 4);
    float*    bc     = (float*)(ws + off);    off = align_up(off + 4 * 4);
    (void)off; (void)ws_size; (void)in_sizes; (void)n_in; (void)out_size;

    // 1) input + weight prep
    {
        const int n = NPOS * 2;
        cvt_f16_kernel<<<(n + 255) / 256, 256, 0, stream>>>(lidar, lidarH, n);
        prep_wT_kernel<<<(32 * 32 + 255) / 256, 256, 0, stream>>>(conv_w[0], wT0, 8, 32);
        prep_wT_kernel<<<(32 * 128 + 255) / 256, 256, 0, stream>>>(conv_w[1], wT1, 128, 128);
        prep_wT_kernel<<<(32 * 128 + 255) / 256, 256, 0, stream>>>(conv_w[2], wT2, 128, 128);
        prep_wT_kernel<<<(32 * 128 + 255) / 256, 256, 0, stream>>>(conv_w[3], wT3, 128, 128);
        collapse_mlp_kernel<<<1, 160, 0, stream>>>(mlp_w1, mlp_b1, mlp_w2, mlp_b2, wc, bc);
    }

    // 2) conv stack (implicit GEMM, WMMA f16->f32); 32 positions per wave
    const int conv_blocks = NPOS / (32 * 8);
    conv2x2_wmma_kernel<2><<<conv_blocks, 256, 0, stream>>>(lidarH, wT0, conv_b[0], fmapA);
    conv2x2_wmma_kernel<32><<<conv_blocks, 256, 0, stream>>>(fmapA, wT1, conv_b[1], fmapB);
    conv2x2_wmma_kernel<32><<<conv_blocks, 256, 0, stream>>>(fmapB, wT2, conv_b[2], fmapA);
    conv2x2_wmma_kernel<32><<<conv_blocks, 256, 0, stream>>>(fmapA, wT3, conv_b[3], fmapB);
    // fmapB now holds context_lidar (f16)

    // 3) encoder GRU
    enc_gru_kernel<<<1, 1024, 0, stream>>>(past, enc_wih, enc_whh, enc_bih, enc_bhh, h_enc);

    // 4) decoder rollout (writes y and logabsdet into d_out)
    dec_rollout_kernel<<<1, 1024, 0, stream>>>(z, past, h_enc, fmapB,
                                               dec_wih, dec_whh, dec_bih, dec_bhh,
                                               wc, bc, out);
}
